// SparseCausalSelfAttention_69818988364518
// MI455X (gfx1250) — matrix operand order repaired and checked by reading, not compiled, not run on hardware
//
#include <hip/hip_runtime.h>

typedef __attribute__((ext_vector_type(16))) __bf16 v16bf;
typedef __attribute__((ext_vector_type(8)))  float  v8f;
typedef __attribute__((ext_vector_type(4)))  unsigned uint32x4;
typedef __attribute__((ext_vector_type(8)))  int      int32x8;
typedef __attribute__((ext_vector_type(4)))  int      int32x4;

#define B_   2
#define T_   4096
#define C_   1024
#define H_   16
#define D_   64
#define WIN_ 256
#define GLB_ 16

union FragB {
    v16bf v;
    uint4 u[2];
};

__device__ inline v8f vzero8() {
    v8f z;
#pragma unroll
    for (int i = 0; i < 8; ++i) z[i] = 0.0f;
    return z;
}

// gfx1250 async global->LDS copy (ASYNCcnt-tracked), 16B per lane.
__device__ inline void async_b128(unsigned lds_addr, unsigned goff, const void* base) {
    asm volatile("global_load_async_to_lds_b128 %0, %1, %2"
                 :
                 : "v"(lds_addr), "v"(goff), "s"(base)
                 : "memory");
}

__device__ inline void wait_async0() {
    asm volatile("s_wait_asynccnt 0" ::: "memory");
}

// ---------------------------------------------------------------------------
// Elementwise fp32 -> bf16 convert
// ---------------------------------------------------------------------------
__global__ void cvt_f32_bf16(const float* __restrict__ in, __bf16* __restrict__ out, int n) {
    int i = blockIdx.x * blockDim.x + threadIdx.x;
    if (i < n) out[i] = (__bf16)in[i];
}

// in: [K][N] fp32 row-major  ->  out: [N][K] bf16 (transposed)
__global__ void transpose_cvt(const float* __restrict__ in, __bf16* __restrict__ out,
                              int K, int N) {
    int i = blockIdx.x * blockDim.x + threadIdx.x;
    if (i >= N * K) return;
    int n = i / K, k = i - n * K;
    out[i] = (__bf16)in[k * N + n];
}

// ---------------------------------------------------------------------------
// bf16 WMMA GEMM: C[M,N] = A[M,K] * Bt[N,K]^T + bias
// A tile: per-lane GLOBAL_LOAD_ASYNC_TO_LDS_B128 (ASYNCcnt)
// B tile: Tensor Data Mover tensor_load_to_lds (TENSORcnt), wave 0 issues;
//         TDM pad feature reproduces the 40-element padded LDS row stride.
// MODE 0: scatter into Q[b,h,t,d], K[b,h,t,d], V^T[b,h,d,t]  (N = 3072)
// MODE 1: fp32 output [M,1024] + bias                        (N = 1024)
// Workgroup tile 128x128, 8 waves (16M x 128N each), BK = 32, double-buffered.
// ---------------------------------------------------------------------------
template <int MODE>
__global__ __launch_bounds__(256) void gemm_bf16(
    const __bf16* __restrict__ A, const __bf16* __restrict__ Bt,
    const float* __restrict__ bias,
    __bf16* __restrict__ Qo, __bf16* __restrict__ Ko, __bf16* __restrict__ Vt,
    float* __restrict__ Out, int Ktot, int Ntot) {
    __shared__ __align__(16) __bf16 As[2][128 * 40];
    __shared__ __align__(16) __bf16 Bs[2][128 * 40];

    const int tid  = threadIdx.x;
    const int wave = tid >> 5;
    const int lane = tid & 31;
    const int lr   = lane & 15;
    const int lh   = lane >> 4;
    const int m0   = blockIdx.x * 128;
    const int n0   = blockIdx.y * 128;

    v8f acc[8];
#pragma unroll
    for (int j = 0; j < 8; ++j) acc[j] = vzero8();

    const int lrow  = tid >> 1;  // 0..127 (tile row for cooperative A loads)
    const int lhalf = tid & 1;   // 0..1   (16-element half of the 32-wide K slab)

    const unsigned gaBase  = (unsigned)(((m0 + lrow) * Ktot + lhalf * 16) * 2);
    const unsigned ldsAoff = (unsigned)(lrow * 40 + lhalf * 16) * 2;

    auto issueA = [&](int k0, int buf) {
        unsigned ga = gaBase + (unsigned)(k0 * 2);
        unsigned la = (unsigned)(unsigned long long)(&As[buf][0]) + ldsAoff;
        async_b128(la, ga, A);
        async_b128(la + 16, ga + 16, A);
    };

    // TDM: move 128(N-rows) x 32(K) bf16 tile of Bt into Bs[buf] with row pad ->
    // LDS stride = 32 data + 8 pad = 40 elements.
    auto issueB = [&](int k0, int buf) {
        unsigned long long gaddr =
            (unsigned long long)Bt + ((unsigned long long)((size_t)n0 * Ktot + k0) * 2ull);
        uint32x4 g0 = {
            1u,                                              // count=1, user D#
            (unsigned)(unsigned long long)(&Bs[buf][0]),     // lds_addr
            (unsigned)gaddr,                                 // global_addr[31:0]
            (unsigned)((gaddr >> 32) & 0x1FFFFFFu) | (2u << 30)  // addr[56:32] | type=2
        };
        int32x8 g1 = {
            (int)((1u << 16) | (1u << 20) | (3u << 22) | (3u << 25)),
            //   data_size=2B | pad_enable | pad_interval=16dw | pad_amount=4dw
            (int)(((unsigned)Ktot & 0xFFFFu) << 16),                       // tensor_dim0 lo
            (int)(((unsigned)Ktot >> 16) | (((unsigned)Ntot & 0xFFFFu) << 16)),  // dim0 hi | dim1 lo
            (int)(((unsigned)Ntot >> 16) | (32u << 16)),                   // dim1 hi | tile_dim0=32
            (int)128,                                                      // tile_dim1=128, tile_dim2=0
            (int)Ktot,                                                     // tensor_dim0_stride lo
            0, 0
        };
        int32x4 gz4 = {0, 0, 0, 0};
        int32x8 gz8 = {0, 0, 0, 0, 0, 0, 0, 0};
        __builtin_amdgcn_tensor_load_to_lds(g0, g1, gz4, gz4, gz8, 0);
    };

    const int nk = Ktot / 32;
    issueA(0, 0);
    if (wave == 0) issueB(0, 0);

    for (int kc = 0; kc < nk; ++kc) {
        wait_async0();                                  // own A-async writes done
        if (wave == 0) __builtin_amdgcn_s_wait_tensorcnt(0);  // B tile landed
        __syncthreads();                                // all waves: tile kc visible
        if (kc + 1 < nk) {
            issueA((kc + 1) * 32, (kc + 1) & 1);
            if (wave == 0) issueB((kc + 1) * 32, (kc + 1) & 1);
        }

        const __bf16* Asb = As[kc & 1];
        const __bf16* Bsb = Bs[kc & 1];

        // A fragment: M = lane%16 (row 16*wave + lr), K = (lane/16)*16 + e
        FragB af;
        {
            const __bf16* base = &Asb[(16 * wave + lr) * 40];
            af.u[0] = *(const uint4*)(base + lh * 16);
            af.u[1] = *(const uint4*)(base + lh * 16 + 8);
        }
#pragma unroll
        for (int j = 0; j < 8; ++j) {
            // B fragment: N = lane%16 (row 16*j + lr of Bt), K = (lane/16)*16 + e
            FragB bf;
            const uint4* bb = (const uint4*)(&Bsb[(16 * j + lr) * 40 + lh * 16]);
            bf.u[0] = bb[0];
            bf.u[1] = bb[1];
            acc[j] = __builtin_amdgcn_wmma_f32_16x16x32_bf16(
                false, af.v, false, bf.v, (short)0, acc[j], false, false);
        }
    }

    // Epilogue: C layout is VGPR v -> row v + 8*(lane/16), col = lane%16
#pragma unroll
    for (int j = 0; j < 8; ++j) {
        int n    = n0 + 16 * j + lr;
        float bv = bias[n];
        if constexpr (MODE == 0) {
            int p  = n >> 10;
            int c  = n & 1023;
            int hh = c >> 6;
            int dd = c & 63;
#pragma unroll
            for (int v = 0; v < 8; ++v) {
                int m  = m0 + 16 * wave + v + 8 * lh;
                int bi = m >> 12;
                int ti = m & 4095;
                __bf16 val = (__bf16)(acc[j][v] + bv);
                size_t bh = (size_t)(bi * H_ + hh);
                if (p == 0)      Qo[(bh * T_ + ti) * D_ + dd] = val;
                else if (p == 1) Ko[(bh * T_ + ti) * D_ + dd] = val;
                else             Vt[(bh * D_ + dd) * T_ + ti] = val;
            }
        } else {
#pragma unroll
            for (int v = 0; v < 8; ++v) {
                int m = m0 + 16 * wave + v + 8 * lh;
                Out[(size_t)m * 1024 + n] = acc[j][v] + bv;
            }
        }
    }
}

// ---------------------------------------------------------------------------
// Sparse-masked flash attention. One wave = one 16-query tile.
// Wave-uniform block skip keeps EXEC effectively all-ones for WMMA.
// Row-sum of P computed with a P @ ones WMMA (replaces shuffle reduction).
// ---------------------------------------------------------------------------
__global__ __launch_bounds__(256) void attn_kernel(
    const __bf16* __restrict__ Q, const __bf16* __restrict__ Kb,
    const __bf16* __restrict__ Vt, __bf16* __restrict__ Y) {
    __shared__ __align__(16) __bf16 P[8][16 * 40];  // per-wave 16x32 P tile

    const int tid  = threadIdx.x;
    const int wave = tid >> 5;
    const int lane = tid & 31;
    const int lr   = lane & 15;
    const int lh   = lane >> 4;
    const int h    = blockIdx.y;
    const int bi   = blockIdx.z;
    const int q0   = (blockIdx.x * 8 + wave) * 16;
    const size_t bh = (size_t)(bi * H_ + h);

    const __bf16* Qp = Q + bh * T_ * D_;
    const __bf16* Kp = Kb + bh * T_ * D_;
    const __bf16* Vp = Vt + bh * D_ * T_;

    // Q fragments for d-chunks 0,1 (A layout)
    FragB qf[2];
    {
        const __bf16* qrow = Qp + (size_t)(q0 + lr) * D_;
#pragma unroll
        for (int f = 0; f < 2; ++f) {
            qf[f].u[0] = *(const uint4*)(qrow + f * 32 + lh * 8);
            qf[f].u[1] = *(const uint4*)(qrow + f * 32 + 16 + lh * 8);
        }
    }

    // Constant all-ones B fragment (for row-sum via WMMA)
    v16bf onev;
#pragma unroll
    for (int i = 0; i < 16; ++i) onev[i] = (__bf16)1.0f;

    v8f O[4];
#pragma unroll
    for (int n = 0; n < 4; ++n) O[n] = vzero8();
    float mrow[8], lsum[8];
#pragma unroll
    for (int v = 0; v < 8; ++v) { mrow[v] = -1e30f; lsum[v] = 0.0f; }

    const float scale = 0.125f;  // 1/sqrt(64)
    const int kcmax = (q0 + 15) / 32;

    for (int kc = 0; kc <= kcmax; ++kc) {
        const int k0 = kc * 32;
        const int hi = q0 + 15 - k0;    // max(i - j) over the 16x32 tile
        const int lo = q0 - (k0 + 31);  // min(i - j)
        // exact block-level mask union (wave-uniform predicate)
        bool any = (k0 < GLB_) || (lo < WIN_) || (((hi >> 7) << 7) >= lo);
        if (!any) continue;

        // S tiles: two 16x16 score tiles over key blocks k0, k0+16 (K-dim = 64)
        v8f S[2];
#pragma unroll
        for (int t2 = 0; t2 < 2; ++t2) {
            const __bf16* krow = Kp + (size_t)(k0 + 16 * t2 + lr) * D_;
            v8f s = vzero8();
#pragma unroll
            for (int f = 0; f < 2; ++f) {
                FragB kf;  // B layout: N = key row (lane%16), K = d run
                kf.u[0] = *(const uint4*)(krow + f * 32 + lh * 8);
                kf.u[1] = *(const uint4*)(krow + f * 32 + 16 + lh * 8);
                s = __builtin_amdgcn_wmma_f32_16x16x32_bf16(
                    false, qf[f].v, false, kf.v, (short)0, s, false, false);
            }
            S[t2] = s;
        }

        // Prefetch next chunk's K/V lines while doing softmax
        if (kc < kcmax) {
            __builtin_prefetch(Kp + (size_t)(k0 + 32 + lr) * D_);
            __builtin_prefetch(Vp + (size_t)(lr)*T_ + k0 + 32);
        }

        // Per-element mask + running max (row = v + 8*lh, col = lr)
        const int j0 = k0 + lr;
        const int j1 = k0 + 16 + lr;
        const bool jg0 = j0 < GLB_;
        const bool jg1 = j1 < GLB_;
        const int dbase = q0 + 8 * lh - k0 - lr;  // (qi - j0) - v
        float s0a[8], s1a[8], alpha[8];
#pragma unroll
        for (int v = 0; v < 8; ++v) {
            int d0 = dbase + v;
            int d1 = d0 - 16;
            bool a0 = (d0 >= 0) && ((d0 < WIN_) || ((d0 & 127) == 0) || jg0);
            bool a1 = (d1 >= 0) && ((d1 < WIN_) || ((d1 & 127) == 0) || jg1);
            float s0 = a0 ? S[0][v] * scale : -1e30f;
            float s1 = a1 ? S[1][v] * scale : -1e30f;
            s0a[v] = s0;
            s1a[v] = s1;
            float m = fmaxf(s0, s1);
#pragma unroll
            for (int off = 1; off < 16; off <<= 1)
                m = fmaxf(m, __shfl_xor(m, off, 32));
            float mn = fmaxf(mrow[v], m);
            alpha[v] = __expf(mrow[v] - mn);
            mrow[v]  = mn;
        }

        // exp, write P (bf16) to per-wave LDS, rescale O rows
        __bf16* Pw = P[wave];
#pragma unroll
        for (int v = 0; v < 8; ++v) {
            float p0 = __expf(s0a[v] - mrow[v]);
            float p1 = __expf(s1a[v] - mrow[v]);
            int prow = v + 8 * lh;
            Pw[prow * 40 + lr]      = (__bf16)p0;
            Pw[prow * 40 + 16 + lr] = (__bf16)p1;
#pragma unroll
            for (int n = 0; n < 4; ++n) O[n][v] *= alpha[v];
        }

        // Re-layout P (16x32) into A fragment (same-wave LDS is in-order)
        FragB pf;
        {
            const __bf16* pb = &Pw[lr * 40];
            pf.u[0] = *(const uint4*)(pb + lh * 16);
            pf.u[1] = *(const uint4*)(pb + lh * 16 + 8);
        }

        // Row-sum of P via WMMA against all-ones B (each lane gets its row's sum)
        v8f lt = __builtin_amdgcn_wmma_f32_16x16x32_bf16(
            false, pf.v, false, onev, (short)0, vzero8(), false, false);
#pragma unroll
        for (int v = 0; v < 8; ++v) lsum[v] = lsum[v] * alpha[v] + lt[v];

        // O += P @ V : 4 d-tiles, V^T rows are contiguous along keys
#pragma unroll
        for (int n = 0; n < 4; ++n) {
            const uint4* vrow =
                (const uint4*)(Vp + (size_t)(n * 16 + lr) * T_ + k0 + lh * 16);
            FragB vf;
            vf.u[0] = vrow[0];
            vf.u[1] = vrow[1];
            O[n] = __builtin_amdgcn_wmma_f32_16x16x32_bf16(
                false, pf.v, false, vf.v, (short)0, O[n], false, false);
        }
    }

    // Normalize and store Y[b*T + t, h*64 + d] as bf16
#pragma unroll
    for (int n = 0; n < 4; ++n) {
#pragma unroll
        for (int v = 0; v < 8; ++v) {
            int row = q0 + v + 8 * lh;
            float val = O[n][v] / lsum[v];
            Y[(size_t)(bi * T_ + row) * C_ + h * 64 + n * 16 + lr] = (__bf16)val;
        }
    }
}

// ---------------------------------------------------------------------------
extern "C" void kernel_launch(void* const* d_in, const int* in_sizes, int n_in,
                              void* d_out, int out_size, void* d_ws, size_t ws_size,
                              hipStream_t stream) {
    (void)in_sizes; (void)n_in; (void)out_size; (void)ws_size;
    const float* x      = (const float*)d_in[0];
    const float* w_qkv  = (const float*)d_in[1];
    const float* b_qkv  = (const float*)d_in[2];
    const float* w_proj = (const float*)d_in[3];
    const float* b_proj = (const float*)d_in[4];
    float* out = (float*)d_out;

    char* ws = (char*)d_ws;
    __bf16* xb     = (__bf16*)(ws);                 // 16 MB  [8192][1024]
    __bf16* wqkvT  = (__bf16*)(ws + 16777216);      //  6 MB  [3072][1024]
    __bf16* wprojT = (__bf16*)(ws + 23068672);      //  2 MB  [1024][1024]
    __bf16* Q      = (__bf16*)(ws + 25165824);      // 16 MB  [b,h,t,d]
    __bf16* Kb     = (__bf16*)(ws + 41943040);      // 16 MB  [b,h,t,d]
    __bf16* Vt     = (__bf16*)(ws + 58720256);      // 16 MB  [b,h,d,t]
    __bf16* Y      = (__bf16*)(ws + 75497472);      // 16 MB  [8192][1024]

    const int NXE = B_ * T_ * C_;  // 8388608
    cvt_f32_bf16<<<NXE / 256, 256, 0, stream>>>(x, xb, NXE);
    transpose_cvt<<<(3072 * 1024) / 256, 256, 0, stream>>>(w_qkv, wqkvT, 1024, 3072);
    transpose_cvt<<<(1024 * 1024) / 256, 256, 0, stream>>>(w_proj, wprojT, 1024, 1024);

    // QKV projection: M=8192, N=3072, K=1024
    gemm_bf16<0><<<dim3(64, 24), 256, 0, stream>>>(xb, wqkvT, b_qkv, Q, Kb, Vt, nullptr,
                                                   1024, 3072);

    // Sparse attention: grid (T/128 q-groups, heads, batch), 8 waves/block
    attn_kernel<<<dim3(T_ / 128, H_, B_), 256, 0, stream>>>(Q, Kb, Vt, Y);

    // Output projection: M=8192, N=1024, K=1024 -> fp32 out + bias
    gemm_bf16<1><<<dim3(64, 8), 256, 0, stream>>>(Y, wprojT, b_proj, nullptr, nullptr, nullptr,
                                                  out, 1024, 1024);
}
